// INFF_ada_28338194219511
// MI455X (gfx1250) — matrix operand order repaired and checked by reading, not compiled, not run on hardware
//
#include <hip/hip_runtime.h>
#include <hip/hip_bf16.h>

// ---------------------------------------------------------------------------
// Types for CDNA5 WMMA (wave32): D = A(16x32 bf16) x B(32x16 bf16) + C(16x16 f32)
// ---------------------------------------------------------------------------
typedef __attribute__((ext_vector_type(16))) __bf16 v16bf;
typedef __attribute__((ext_vector_type(8)))  float  v8f;
typedef __attribute__((ext_vector_type(4)))  unsigned u32x4;
typedef __attribute__((ext_vector_type(8)))  int      i32x8;
typedef __attribute__((ext_vector_type(4)))  int      i32x4;

struct Frag16 {
    union {
        v16bf v;
        unsigned short u[16];
    };
};

__device__ __forceinline__ unsigned short f2bf(float f) {
    unsigned u = __float_as_uint(f);
    unsigned r = (u + 0x7FFFu + ((u >> 16) & 1u)) >> 16;   // round-to-nearest-even
    return (unsigned short)r;
}

// A-matrix (16x32, bf16) fragment from an LDS tile, contiguous k = 16h + s, same order as load_b_frag
__device__ __forceinline__ void load_a_frag(Frag16& f, const unsigned short* lds,
                                            int row0, int k0, int stride, int lane) {
    const int h = lane >> 4, m = lane & 15;
    const unsigned short* base = lds + (size_t)(row0 + m) * stride + k0;
#pragma unroll
    for (int v = 0; v < 8; ++v) {
        int k = 16 * h + 2 * v;
        f.u[2 * v]     = base[k];
        f.u[2 * v + 1] = base[k + 1];
    }
}

// B-matrix (32x16, bf16): lanes 0-15 K=0..15, lanes 16-31 K=16..31,
// VGPR v -> K = 16h + 2v (+1), column = col0 + (lane & 15)
__device__ __forceinline__ void load_b_frag(Frag16& f, const unsigned short* lds,
                                            int col0, int stride, int lane) {
    const int h = lane >> 4, n = lane & 15;
#pragma unroll
    for (int v = 0; v < 8; ++v) {
        int k = 16 * h + 2 * v;
        f.u[2 * v]     = lds[(size_t)k * stride + col0 + n];
        f.u[2 * v + 1] = lds[(size_t)(k + 1) * stride + col0 + n];
    }
}

__device__ __forceinline__ v8f wmma_bf16(const Frag16& a, const Frag16& b, v8f c) {
    return __builtin_amdgcn_wmma_f32_16x16x32_bf16(
        false, a.v, false, b.v, (short)0, c, false, false);
}

// ---------------------------------------------------------------------------
// CDNA5 async global->LDS copies (ASYNCcnt) via inline asm
// ---------------------------------------------------------------------------
__device__ __forceinline__ void async_b64(unsigned lds_off, const void* g) {
    asm volatile("global_load_async_to_lds_b64 %0, %1, off"
                 :: "v"(lds_off), "v"(g) : "memory");
}
__device__ __forceinline__ void async_b128(unsigned lds_off, const void* g) {
    asm volatile("global_load_async_to_lds_b128 %0, %1, off"
                 :: "v"(lds_off), "v"(g) : "memory");
}
__device__ __forceinline__ void wait_async0() {
    asm volatile("s_wait_asynccnt 0x0" ::: "memory");
}
__device__ __forceinline__ unsigned lds_off(const void* p) {
    return (unsigned)(size_t)p;   // low 32 bits of a generic shared pointer = LDS offset
}

// ---------------------------------------------------------------------------
// Tensor Data Mover: 1-D contiguous global->LDS copy of nelem16 bf16 elements.
// ---------------------------------------------------------------------------
#if __has_builtin(__builtin_amdgcn_tensor_load_to_lds)
#define HAS_TDM 1
__device__ __forceinline__ void tdm_load_1d(void* lds, const void* gsrc, unsigned nelem16) {
    const unsigned long long ga = (unsigned long long)gsrc;
    u32x4 g0;
    g0[0] = 1u;                                        // count=1, user descriptor
    g0[1] = lds_off(lds);                              // lds_addr (bytes)
    g0[2] = (unsigned)(ga & 0xFFFFFFFFu);              // global_addr[31:0]
    g0[3] = (unsigned)((ga >> 32) & 0x01FFFFFFu)       // global_addr[56:32]
          | 0x80000000u;                               // type=2 ("image")
    i32x8 g1 = {};
    g1[0] = 0x00010000;                                // data_size=1 (2B), no flags
    g1[1] = (int)((nelem16 & 0xFFFFu) << 16);          // tensor_dim0[15:0]
    g1[2] = (int)(((nelem16 >> 16) & 0xFFFFu)          // tensor_dim0[31:16]
          | (1u << 16));                               // tensor_dim1 = 1
    g1[3] = (int)((nelem16 & 0xFFFFu) << 16);          // tile_dim0 = nelem16
    g1[4] = 1;                                         // tile_dim1 = 1
    g1[5] = (int)nelem16;                              // tensor_dim0_stride[31:0]
    g1[6] = (int)((nelem16 & 0xFFFFu) << 16);          // tensor_dim1_stride[15:0]
    g1[7] = (int)((nelem16 >> 16) & 0xFFFFu);          // tensor_dim1_stride[47:16]
    i32x4 z = {};
#if defined(__clang_major__) && (__clang_major__ >= 23)
    i32x8 z8 = {};
    __builtin_amdgcn_tensor_load_to_lds(g0, g1, z, z, z8, 0);
#else
    __builtin_amdgcn_tensor_load_to_lds(g0, g1, z, z, 0);
#endif
}
#else
#define HAS_TDM 0
#endif

// ---------------------------------------------------------------------------
// Problem constants
// ---------------------------------------------------------------------------
#define NB   64
#define LSP  1024
#define FQ   513
#define FQP  544           // FQ padded to 17 tiles of 32
#define DD   256
#define OMEGA 30.0f
#define TWO_PI 6.28318530717958647692f
#define INV_SQRT_L 0.03125f  // 1/sqrt(1024)

// ---------------------------------------------------------------------------
// Small f32 kernels (batch-invariant prologue)
// ---------------------------------------------------------------------------
__global__ void k_siren_first(const float* __restrict__ sw0, const float* __restrict__ sb0,
                              float* __restrict__ h) {
    const int l = blockIdx.x, j = threadIdx.x;
    const float t = -1.0f + 2.0f * (float)l / 1023.0f;
    h[l * DD + j] = sinf(OMEGA * (t * sw0[j] + sb0[j]));
}

__global__ void k_siren_mid(const float* __restrict__ in, const float* __restrict__ w,
                            const float* __restrict__ b, float* __restrict__ out, int act) {
    const int l = blockIdx.x, j = threadIdx.x;
    float acc = b[j];
    for (int k = 0; k < DD; ++k) acc = fmaf(in[l * DD + k], w[k * DD + j], acc);
    out[l * DD + j] = act ? sinf(OMEGA * acc) : acc;
}

__global__ void k_ln(const float* __restrict__ phi, const float* __restrict__ g,
                     const float* __restrict__ be, float* __restrict__ o) {
    __shared__ float s1[DD], s2[DD];
    const int l = blockIdx.x, j = threadIdx.x;
    const float x = phi[l * DD + j];
    s1[j] = x; s2[j] = x * x;
    __syncthreads();
    for (int st = 128; st > 0; st >>= 1) {
        if (j < st) { s1[j] += s1[j + st]; s2[j] += s2[j + st]; }
        __syncthreads();
    }
    const float m = s1[0] * (1.0f / DD);
    const float v = s2[0] * (1.0f / DD) - m * m;
    o[l * DD + j] = (x - m) * rsqrtf(v + 1e-5f) * g[j] + be[j];
}

__global__ void k_bcast_fx(const float* __restrict__ src, float* __restrict__ dst) {
    const size_t i = (size_t)blockIdx.x * 256 + threadIdx.x;   // covers B*L*D
    dst[i] = src[i & (size_t)(LSP * DD - 1)];                  // L*D = 2^18
}

// DFT basis (bf16): Cb = cos(2*pi*q*l/L)/32 ; Sb = -sin(2*pi*q*l/L)/32
__global__ void k_basis(unsigned short* __restrict__ Cb, unsigned short* __restrict__ Sb) {
    const int idx = blockIdx.x * 256 + threadIdx.x;            // FQP*L threads
    const int q = idx >> 10, l = idx & 1023;
    const int m = (q * l) & 1023;                              // exact phase mod L
    const float ang = -TWO_PI * (float)m * (1.0f / 1024.0f);
    float s, c;
    __sincosf(ang, &s, &c);
    Cb[idx] = f2bf(c * INV_SQRT_L);
    Sb[idx] = f2bf(s * INV_SQRT_L);
}

// f = rfft(fxln)*scale + bias  (batch-invariant), incremental-rotation DFT
__global__ void k_fvec(const float* __restrict__ fx,
                       const float* __restrict__ sre, const float* __restrict__ sim,
                       const float* __restrict__ bre, const float* __restrict__ bim,
                       float* __restrict__ fvr, float* __restrict__ fvi) {
    const int q = blockIdx.x, d = threadIdx.x;
    const float wang = TWO_PI * (float)q * (1.0f / 1024.0f);
    float cb, sb;
    __sincosf(wang, &sb, &cb);
    float c = 1.0f, s = 0.0f, ar = 0.0f, ai = 0.0f;
    for (int l = 0; l < LSP; ++l) {
        const float x = fx[l * DD + d];
        ar = fmaf(x, c, ar);
        ai = fmaf(x, -s, ai);
        const float cn = c * cb - s * sb;
        s = c * sb + s * cb;
        c = cn;
    }
    ar *= INV_SQRT_L; ai *= INV_SQRT_L;
    const size_t o = (size_t)q * DD + d;
    fvr[o] = ar * sre[o] - ai * sim[o] + bre[o];
    fvi[o] = ar * sim[o] + ai * sre[o] + bim[o];
}

// U = f @ cvw1[0:D, :]   (batch-invariant half of the concat GEMM)
__global__ void k_umat(const float* __restrict__ fvr, const float* __restrict__ fvi,
                       const float* __restrict__ cvw1, float* __restrict__ Ur,
                       float* __restrict__ Ui) {
    const int q = blockIdx.x, j = threadIdx.x;
    float ar = 0.0f, ai = 0.0f;
    for (int k = 0; k < DD; ++k) {
        const float w = cvw1[k * DD + j];
        ar = fmaf(fvr[q * DD + k], w, ar);
        ai = fmaf(fvi[q * DD + k], w, ai);
    }
    Ur[q * DD + j] = ar;
    Ui[q * DD + j] = ai;
}

__global__ void k_f2bf(const float* __restrict__ in, unsigned short* __restrict__ out, int n) {
    const int i = blockIdx.x * 256 + threadIdx.x;
    if (i < n) out[i] = f2bf(in[i]);
}

// ---------------------------------------------------------------------------
// Heavy kernel 1: batched rDFT of x_cond as bf16 WMMA GEMM.
// Staging is fully async (GLOBAL_LOAD_ASYNC_TO_LDS_*, ASYNCcnt).
// ---------------------------------------------------------------------------
__global__ __launch_bounds__(256) void k_dft_wmma(const unsigned short* __restrict__ Cb,
                                                  const unsigned short* __restrict__ Sb,
                                                  const unsigned short* __restrict__ xcb,
                                                  unsigned short* __restrict__ fcr,
                                                  unsigned short* __restrict__ fci) {
    __shared__ unsigned short Cs[32][36];
    __shared__ unsigned short Ss[32][36];
    __shared__ unsigned short Xs[32][72];

    const int tid  = threadIdx.x;
    const int lane = tid & 31;
    const int w    = tid >> 5;
    const int q0   = blockIdx.x * 32;
    const int n0   = blockIdx.y * 64;
    const int b    = blockIdx.z;
    const int sm   = (w & 1) * 16;
    const int sn   = (w >> 1) * 16;

    v8f accr = {};
    v8f acci = {};

    const int arow = tid >> 3, acol = (tid & 7) * 4;   // 32 rows x 4 ushorts
    const int xrow = tid >> 3, xcol = (tid & 7) * 8;   // 32 rows x 8 ushorts
    const unsigned c_lds = lds_off(&Cs[arow][acol]);
    const unsigned s_lds = lds_off(&Ss[arow][acol]);
    const unsigned x_lds = lds_off(&Xs[xrow][xcol]);

    for (int k0 = 0; k0 < LSP; k0 += 32) {
        __syncthreads();
        const size_t aoff = (size_t)(q0 + arow) * LSP + k0 + acol;
        const size_t xoff = ((size_t)b * LSP + k0 + xrow) * DD + n0 + xcol;
        async_b64(c_lds, Cb + aoff);
        async_b64(s_lds, Sb + aoff);
        async_b128(x_lds, xcb + xoff);
        if (k0 + 32 < LSP)   // pull the next x tile toward the caches
            __builtin_prefetch(xcb + xoff + (size_t)32 * DD, 0, 0);
        wait_async0();
        __syncthreads();

        Frag16 ac, as, bx;
        load_a_frag(ac, &Cs[0][0], sm, 0, 36, lane);
        load_a_frag(as, &Ss[0][0], sm, 0, 36, lane);
        load_b_frag(bx, &Xs[0][0], sn, 72, lane);
        accr = wmma_bf16(ac, bx, accr);
        acci = wmma_bf16(as, bx, acci);
    }

    const int h = lane >> 4, nn = lane & 15;
    const int d = n0 + sn + nn;
#pragma unroll
    for (int j = 0; j < 8; ++j) {
        const int q = q0 + sm + j + 8 * h;
        if (q < FQ) {
            const size_t o = ((size_t)b * FQ + q) * DD + d;
            fcr[o] = f2bf(accr[j]);
            fci[o] = f2bf(acci[j]);
        }
    }
}

// ---------------------------------------------------------------------------
// Heavy kernel 2: fused  T = relu(U + fc @ W1_bot) ;  Z = T @ W2 ; complex epilogue.
// A-tiles staged async; 16KB weight tiles staged by the Tensor Data Mover.
// ---------------------------------------------------------------------------
__global__ __launch_bounds__(256) void k_mlp_wmma(const unsigned short* __restrict__ fcr,
                                                  const unsigned short* __restrict__ fci,
                                                  const unsigned short* __restrict__ w1b,
                                                  const unsigned short* __restrict__ w2b,
                                                  const float* __restrict__ Ur,
                                                  const float* __restrict__ Ui,
                                                  const float* __restrict__ x_re,
                                                  const float* __restrict__ x_im,
                                                  float* __restrict__ out,
                                                  float* __restrict__ f_ri) {
    __shared__ unsigned short Ar[32][36];
    __shared__ unsigned short Ai[32][36];
    __shared__ unsigned short Bs[32][256];   // unpadded: one contiguous 16KB TDM tile
    __shared__ unsigned short Tr[32][264];
    __shared__ unsigned short Ti[32][264];

    const int tid  = threadIdx.x;
    const int lane = tid & 31;
    const int w    = tid >> 5;
    const int q0   = blockIdx.x * 32;
    const int b    = blockIdx.y;
    const int sm   = (w & 1) * 16;
    const int snb  = (w >> 1) * 64;

    const int arow = tid >> 3, acol = (tid & 7) * 4;
    const int wrow = tid >> 5, wcol = (tid & 31) * 8;
    const unsigned ar_lds = lds_off(&Ar[arow][acol]);
    const unsigned ai_lds = lds_off(&Ai[arow][acol]);

    // ---- stage 1: fc @ W1_bot -------------------------------------------
    v8f accR[4] = {};
    v8f accI[4] = {};
    for (int kk = 0; kk < 8; ++kk) {
        const int k0 = kk * 32;
        __syncthreads();
        {
            const int q = q0 + arow;
            if (q < FQ) {
                const size_t o = ((size_t)b * FQ + q) * DD + k0 + acol;
                async_b64(ar_lds, fcr + o);
                async_b64(ai_lds, fci + o);
            } else {
                uint2 z; z.x = 0u; z.y = 0u;
                *(uint2*)&Ar[arow][acol] = z;
                *(uint2*)&Ai[arow][acol] = z;
            }
#if HAS_TDM
            if (w == 0) {
                tdm_load_1d(&Bs[0][0], w1b + (size_t)k0 * DD, 32 * DD);
                __builtin_amdgcn_s_wait_tensorcnt(0);
            }
#else
#pragma unroll
            for (int rr = 0; rr < 4; ++rr) {
                const int row = wrow + rr * 8;
                *(uint4*)&Bs[row][wcol] = *(const uint4*)(w1b + (size_t)(k0 + row) * DD + wcol);
            }
#endif
        }
        wait_async0();
        __syncthreads();

        Frag16 far, fai;
        load_a_frag(far, &Ar[0][0], sm, 0, 36, lane);
        load_a_frag(fai, &Ai[0][0], sm, 0, 36, lane);
#pragma unroll
        for (int t = 0; t < 4; ++t) {
            Frag16 fb;
            load_b_frag(fb, &Bs[0][0], snb + 16 * t, 256, lane);
            accR[t] = wmma_bf16(far, fb, accR[t]);
            accI[t] = wmma_bf16(fai, fb, accI[t]);
        }
    }
    __syncthreads();

    // ---- epilogue 1: add hoisted U, relu, stash T as bf16 in LDS --------
    {
        const int h = lane >> 4, nn = lane & 15;
#pragma unroll
        for (int t = 0; t < 4; ++t) {
            const int col = snb + 16 * t + nn;
#pragma unroll
            for (int j = 0; j < 8; ++j) {
                const int row = sm + j + 8 * h;
                const int q = q0 + row;
                float ur = 0.0f, ui = 0.0f;
                if (q < FQ) {
                    ur = Ur[(size_t)q * DD + col];
                    ui = Ui[(size_t)q * DD + col];
                }
                Tr[row][col] = f2bf(fmaxf(accR[t][j] + ur, 0.0f));
                Ti[row][col] = f2bf(fmaxf(accI[t][j] + ui, 0.0f));
            }
        }
    }
    __syncthreads();

    // ---- stage 2: T @ W2 -------------------------------------------------
    v8f zR[4] = {};
    v8f zI[4] = {};
    for (int kk = 0; kk < 8; ++kk) {
        const int k0 = kk * 32;
        __syncthreads();   // protect Bs overwrite
#if HAS_TDM
        if (w == 0) {
            tdm_load_1d(&Bs[0][0], w2b + (size_t)k0 * DD, 32 * DD);
            __builtin_amdgcn_s_wait_tensorcnt(0);
        }
#else
#pragma unroll
        for (int rr = 0; rr < 4; ++rr) {
            const int row = wrow + rr * 8;
            *(uint4*)&Bs[row][wcol] = *(const uint4*)(w2b + (size_t)(k0 + row) * DD + wcol);
        }
#endif
        __syncthreads();

        Frag16 far, fai;
        load_a_frag(far, &Tr[0][0], sm, k0, 264, lane);
        load_a_frag(fai, &Ti[0][0], sm, k0, 264, lane);
#pragma unroll
        for (int t = 0; t < 4; ++t) {
            Frag16 fb;
            load_b_frag(fb, &Bs[0][0], snb + 16 * t, 256, lane);
            zR[t] = wmma_bf16(far, fb, zR[t]);
            zI[t] = wmma_bf16(fai, fb, zI[t]);
        }
    }

    // ---- epilogue 2: complex modulation, write out + f_ri ---------------
    {
        const int h = lane >> 4, nn = lane & 15;
#pragma unroll
        for (int t = 0; t < 4; ++t) {
            const int d = snb + 16 * t + nn;
#pragma unroll
            for (int j = 0; j < 8; ++j) {
                const int q = q0 + sm + j + 8 * h;
                if (q >= FQ) continue;
                const size_t o = ((size_t)b * FQ + q) * DD + d;
                const float fr = zR[t][j], fi = zI[t][j];
                f_ri[o * 2]     = fr;
                f_ri[o * 2 + 1] = fi;
                const float xr = x_re[o], xi = x_im[o];
                out[o * 2]     = xr * fr - xi * fi;
                out[o * 2 + 1] = xr * fi + xi * fr;
            }
        }
    }
}

// ---------------------------------------------------------------------------
// Host-side launch
// ---------------------------------------------------------------------------
extern "C" void kernel_launch(void* const* d_in, const int* in_sizes, int n_in,
                              void* d_out, int out_size, void* d_ws, size_t ws_size,
                              hipStream_t stream) {
    (void)in_sizes; (void)n_in; (void)out_size; (void)ws_size;

    const float* x_re   = (const float*)d_in[0];
    const float* x_im   = (const float*)d_in[1];
    const float* x_cond = (const float*)d_in[2];
    // d_in[3] = x_cond2 (ignored: adaptive LN is off in the reference)
    const float* sw0 = (const float*)d_in[4];
    const float* sb0 = (const float*)d_in[5];
    const float* sw1 = (const float*)d_in[6];
    const float* sb1 = (const float*)d_in[7];
    const float* sw2 = (const float*)d_in[8];
    const float* sb2 = (const float*)d_in[9];
    const float* sw3 = (const float*)d_in[10];
    const float* sb3 = (const float*)d_in[11];
    const float* gamma = (const float*)d_in[12];
    const float* beta  = (const float*)d_in[13];
    const float* scale_re = (const float*)d_in[14];
    const float* scale_im = (const float*)d_in[15];
    const float* bias_re  = (const float*)d_in[16];
    const float* bias_im  = (const float*)d_in[17];
    const float* cvw1 = (const float*)d_in[18];   // (512, 256)
    const float* cvw2 = (const float*)d_in[19];   // (256, 256)

    float* out0 = (float*)d_out;                                   // (B,FQ,D,2)
    float* out1 = out0 + (size_t)NB * FQ * DD * 2;                 // f_ri
    float* out2 = out1 + (size_t)NB * FQ * DD * 2;                 // f_x (B,L,D)

    // ---- workspace carve-up ---------------------------------------------
    char* p = (char*)d_ws;
    auto carve = [&p](size_t bytes) -> void* {
        void* r = (void*)p;
        p += (bytes + 255) & ~(size_t)255;
        return r;
    };
    float* hA   = (float*)carve((size_t)LSP * DD * 4);
    float* hB   = (float*)carve((size_t)LSP * DD * 4);
    float* fxln = (float*)carve((size_t)LSP * DD * 4);
    float* fvr  = (float*)carve((size_t)FQ * DD * 4);
    float* fvi  = (float*)carve((size_t)FQ * DD * 4);
    float* Ur   = (float*)carve((size_t)FQ * DD * 4);
    float* Ui   = (float*)carve((size_t)FQ * DD * 4);
    unsigned short* Cb  = (unsigned short*)carve((size_t)FQP * LSP * 2);
    unsigned short* Sb  = (unsigned short*)carve((size_t)FQP * LSP * 2);
    unsigned short* w1b = (unsigned short*)carve((size_t)DD * DD * 2);
    unsigned short* w2b = (unsigned short*)carve((size_t)DD * DD * 2);
    unsigned short* xcb = (unsigned short*)carve((size_t)NB * LSP * DD * 2);
    unsigned short* fcr = (unsigned short*)carve((size_t)NB * FQ * DD * 2);
    unsigned short* fci = (unsigned short*)carve((size_t)NB * FQ * DD * 2);

    // ---- batch-invariant prologue ---------------------------------------
    k_siren_first<<<LSP, DD, 0, stream>>>(sw0, sb0, hA);
    k_siren_mid<<<LSP, DD, 0, stream>>>(hA, sw1, sb1, hB, 1);
    k_siren_mid<<<LSP, DD, 0, stream>>>(hB, sw2, sb2, hA, 1);
    k_siren_mid<<<LSP, DD, 0, stream>>>(hA, sw3, sb3, hB, 0);     // phi
    k_ln<<<LSP, DD, 0, stream>>>(hB, gamma, beta, fxln);
    k_bcast_fx<<<(NB * LSP * DD) / 256, 256, 0, stream>>>(fxln, out2);
    k_basis<<<(FQP * LSP) / 256, 256, 0, stream>>>(Cb, Sb);
    k_fvec<<<FQ, DD, 0, stream>>>(fxln, scale_re, scale_im, bias_re, bias_im, fvr, fvi);
    k_umat<<<FQ, DD, 0, stream>>>(fvr, fvi, cvw1, Ur, Ui);

    // ---- bf16 conversions ------------------------------------------------
    k_f2bf<<<(DD * DD) / 256, 256, 0, stream>>>(cvw1 + (size_t)DD * DD, w1b, DD * DD);
    k_f2bf<<<(DD * DD) / 256, 256, 0, stream>>>(cvw2, w2b, DD * DD);
    k_f2bf<<<(NB * LSP * DD) / 256, 256, 0, stream>>>(x_cond, xcb, NB * LSP * DD);

    // ---- heavy WMMA stages ----------------------------------------------
    k_dft_wmma<<<dim3(FQP / 32, DD / 64, NB), 256, 0, stream>>>(Cb, Sb, xcb, fcr, fci);
    k_mlp_wmma<<<dim3(FQP / 32, NB), 256, 0, stream>>>(fcr, fci, w1b, w2b, Ur, Ui,
                                                       x_re, x_im, out0, out1);
}
